// _IPEXLlamaAttention_40329742909532
// MI455X (gfx1250) — matrix operand order repaired and checked by reading, not compiled, not run on hardware
//
#include <hip/hip_runtime.h>
#include <hip/hip_bf16.h>

// ---------------------------------------------------------------------------
// Llama attention block on gfx1250 (MI455X).
// Pipeline: fp32->bf16 pre-pass (bandwidth-bound, once) ->
//           LDS-staged, double-buffered WMMA GEMMs with fused epilogues
//           (RoPE / V-transpose / residual) ->
//           transposed WMMA flash attention.
// All matmuls via v_wmma_f32_16x16x32_bf16 (wave32, fp32 accumulate).
// ---------------------------------------------------------------------------

typedef __attribute__((ext_vector_type(16))) __bf16 v16bf;
typedef __attribute__((ext_vector_type(8)))  __bf16 v8bf;
typedef __attribute__((ext_vector_type(8)))  float  v8f;
typedef __attribute__((ext_vector_type(4)))  float  v4f;

#define SEQL   2048
#define HID    4096
#define NHEAD  32
#define NKVH   8
#define HDIM   128

#define BROW   40              // padded LDS row stride (bf16 elems) for 32 k

// Epilogue modes for the GEMM
#define EPI_ROPE 0   // apply RoPE pairs in-register, write bf16 [head][seq][128]
#define EPI_VT   1   // write bf16 transposed  [kvh][d][seq]
#define EPI_RES  2   // add fp32 residual, write fp32 row-major (o_proj)

__device__ __forceinline__ v8f wmma_bf16(v16bf a, v16bf b, v8f c) {
  return __builtin_amdgcn_wmma_f32_16x16x32_bf16(
      /*neg_a=*/false, a, /*neg_b=*/false, b,
      /*c_mod=*/(short)0, c, /*reuse_a=*/false, /*reuse_b=*/false);
}

__device__ __forceinline__ v16bf cat8(v8bf lo, v8bf hi) {
  return __builtin_shufflevector(lo, hi, 0,1,2,3,4,5,6,7,8,9,10,11,12,13,14,15);
}

// A-operand (16-bit, 16xK=32): lane m=lane%16, half=lane/16 holds
// k = {off..off+7, off+16..off+23} with off = half*8. Two 16B loads.
__device__ __forceinline__ v16bf load_a_bf16(const __bf16* p) {
  return cat8(*(const v8bf*)(p), *(const v8bf*)(p + 16));
}

// ===================== fp32 -> bf16 conversion pre-pass =====================
__global__ __launch_bounds__(256)
void cvt_f32_to_bf16(const float* __restrict__ X, __bf16* __restrict__ Y, int n8)
{
  int t = blockIdx.x * blockDim.x + threadIdx.x;
  if (t >= n8) return;
  const v4f* xp = (const v4f*)X + (size_t)t * 2;
  v4f a = xp[0], b = xp[1];
  v8bf y;
#pragma unroll
  for (int e = 0; e < 4; ++e) {
    y[e]     = (__bf16)a[e];
    y[4 + e] = (__bf16)b[e];
  }
  *((v8bf*)Y + t) = y;
}

// =============================== GEMM ======================================
// C[M,N] = A[M,K] * W[N,K]^T, bf16 operands, fp32 accumulate.
// Block (8 waves) owns a 128M x 128N tile. The 128N x 32K weight slab is
// staged once per block into double-buffered LDS (8-way reuse -> 8x less L2
// traffic for W); each wave computes a 16M x 128N strip, reading B fragments
// from LDS and its private A fragment from global (L2).
template <int EPI>
__global__ __launch_bounds__(256)
void wmma_gemm_xwT(const __bf16* __restrict__ A, const __bf16* __restrict__ W,
                   const float* __restrict__ res, void* __restrict__ outv,
                   int M, int N, int K)
{
  __shared__ alignas(32) __bf16 sB[2][128 * BROW];

  const int lane = threadIdx.x & 31;
  const int bw   = threadIdx.x >> 5;        // wave in block (0..7)
  const int nTiles = N >> 7;                // N/128
  const int mb = blockIdx.x / nTiles;       // 128-row M block
  const int nt = blockIdx.x % nTiles;
  const int m0 = (mb << 7) + (bw << 4);     // this wave's 16-row strip
  const int n0 = nt << 7;
  const int half = lane >> 4;
  const int ln   = lane & 15;

  v8f c[8];
#pragma unroll
  for (int j = 0; j < 8; ++j)
#pragma unroll
    for (int e = 0; e < 8; ++e) c[j][e] = 0.0f;

  const __bf16* arow = A + (size_t)(m0 + ln) * K;

  // Cooperative B staging: thread t covers row n_l = t/2, k-half kh = t%2.
  const int n_l = threadIdx.x >> 1;
  const int kh  = threadIdx.x & 1;
  const __bf16* gB   = W + (size_t)(n0 + n_l) * K + kh * 16;
  __bf16*       lB0  = &sB[0][n_l * BROW + kh * 16];
  __bf16*       lB1  = &sB[1][n_l * BROW + kh * 16];

  // Prologue: stage k0 = 0 into buffer 0.
  {
    v8bf r0 = *(const v8bf*)(gB);
    v8bf r1 = *(const v8bf*)(gB + 8);
    *(v8bf*)(lB0)     = r0;
    *(v8bf*)(lB0 + 8) = r1;
  }
  __syncthreads();

  int cur = 0;
  for (int k0 = 0; k0 < K; k0 += 32) {
    const bool more = (k0 + 32) < K;
    // Fetch next B slab into registers while computing on the current one.
    v8bf r0, r1;
    if (more) {
      r0 = *(const v8bf*)(gB + k0 + 32);
      r1 = *(const v8bf*)(gB + k0 + 40);
    }

    // A fragment: contiguous k layout (16 k at 16*half), one load from global (L2).
    v16bf a = *(const v16bf*)(arow + k0 + half * 16);

    // B fragments from LDS: column n == staged row, 16 contiguous k.
    const __bf16* bbase = &sB[cur][half * 16];
#pragma unroll
    for (int j = 0; j < 8; ++j) {
      const __bf16* bp = bbase + (j * 16 + ln) * BROW;
      v16bf b = cat8(*(const v8bf*)(bp), *(const v8bf*)(bp + 8));
      c[j] = wmma_bf16(a, b, c[j]);
    }

    // Store next slab into the other buffer, then one barrier per k-step.
    if (more) {
      __bf16* ld = cur ? lB0 : lB1;
      *(v8bf*)(ld)     = r0;
      *(v8bf*)(ld + 8) = r1;
    }
    __syncthreads();
    cur ^= 1;
  }

  // ---- Epilogues. C layout: lane = n (within 16), vgpr i -> m = i + 8*half.
  if (EPI == EPI_ROPE) {
    // nt == head index (N-tile width == HDIM). Pair (d, d+64) = (c[j], c[j+4]).
    __bf16* Oh = (__bf16*)outv + (size_t)nt * SEQL * HDIM;
#pragma unroll
    for (int j = 0; j < 4; ++j) {
      const int d = j * 16 + ln;                       // 0..63
      const float inv = __expf(-(float)d * (9.210340371976184f / 64.0f));
#pragma unroll
      for (int i = 0; i < 8; ++i) {
        const int m = m0 + i + half * 8;               // position id
        const float ang = (float)m * inv;
        const float cs = __cosf(ang), sn = __sinf(ang);
        const float x1 = c[j][i], x2 = c[j + 4][i];
        __bf16* orow = Oh + (size_t)m * HDIM;
        orow[d]      = (__bf16)(x1 * cs - x2 * sn);
        orow[d + 64] = (__bf16)(x2 * cs + x1 * sn);
      }
    }
  } else if (EPI == EPI_VT) {
    // nt == kv head; write transposed: Vt[(kvh*128 + d) * SEQL + q]
    __bf16* Vt = (__bf16*)outv;
#pragma unroll
    for (int j = 0; j < 8; ++j) {
      const int d = j * 16 + ln;
#pragma unroll
      for (int i = 0; i < 8; ++i) {
        const int q = m0 + i + half * 8;
        Vt[((size_t)nt * HDIM + d) * SEQL + q] = (__bf16)c[j][i];
      }
    }
  } else { // EPI_RES: o_proj + residual, fp32 out
    float* out = (float*)outv;
#pragma unroll
    for (int j = 0; j < 8; ++j) {
      const int n = n0 + j * 16 + ln;
#pragma unroll
      for (int i = 0; i < 8; ++i) {
        const int m = m0 + i + half * 8;
        out[(size_t)m * N + n] = c[j][i] + res[(size_t)m * N + n];
      }
    }
  }
}

// ============================ Flash attention ==============================
// One wave per (head, 16-row q block). Works in transposed space:
//   S^T = K * Q^T   (both operands d-contiguous)
//   O^T = V^T * P^T (P^T comes straight out of the S^T accumulator layout)
// q index lives in the lane dimension -> softmax stats are per-lane,
// O^T rescale is a pure per-lane multiply.
__global__ __launch_bounds__(256)
void flash_attn_wmma(const __bf16* __restrict__ Qb, const __bf16* __restrict__ Kb,
                     const __bf16* __restrict__ Vt, __bf16* __restrict__ AttnO)
{
  const int lane = threadIdx.x & 31;
  const int wid  = (int)((blockIdx.x * blockDim.x + threadIdx.x) >> 5);
  if (wid >= NHEAD * (SEQL / 16)) return;
  const int h   = wid >> 7;            // / (SEQL/16)
  const int qb  = wid & 127;
  const int q0  = qb << 4;
  const int kvh = h >> 2;              // GQA: 4 Q heads per KV head
  const int half = lane >> 4;
  const int ln   = lane & 15;

  const __bf16* Qh = Qb + (size_t)h   * SEQL * HDIM;
  const __bf16* Kh = Kb + (size_t)kvh * SEQL * HDIM;
  const __bf16* Vh = Vt + (size_t)kvh * HDIM * SEQL;

  // Q as B-operand, resident for the whole kv loop:
  // lane holds column q=q0+ln, 16 contiguous d at 32c + 16*half.
  v16bf bq[4];
#pragma unroll
  for (int c = 0; c < 4; ++c)
    bq[c] = *(const v16bf*)(Qh + (size_t)(q0 + ln) * HDIM + c * 32 + half * 16);

  float m_i = -1e30f, l_i = 0.0f;
  v8f o[8];
#pragma unroll
  for (int j = 0; j < 8; ++j)
#pragma unroll
    for (int e = 0; e < 8; ++e) o[j][e] = 0.0f;

  const float rsc = 0.08838834764831845f;  // 1/sqrt(128)
  const int qg   = q0 + ln;
  const int nkv  = (q0 + 16 + 31) >> 5;    // # of 32-wide kv chunks (causal)

  for (int kc = 0; kc < nkv; ++kc) {
    const int kv0 = kc << 5;

    // ---- S^T fragments: two 16-kv slabs, accumulate over d=128
    v8f st[2];
#pragma unroll
    for (int s = 0; s < 2; ++s) {
#pragma unroll
      for (int e = 0; e < 8; ++e) st[s][e] = 0.0f;
      const int krow = kv0 + s * 16 + ln;
#pragma unroll
      for (int c = 0; c < 4; ++c) {
        v16bf ak = *(const v16bf*)(Kh + (size_t)krow * HDIM + c * 32 + half * 16);
        st[s] = wmma_bf16(ak, bq[c], st[s]);
      }
    }

    // ---- scale, causal mask, online softmax (kv = vgpr i + 8*half [+16])
    float mloc = -1e30f;
    float p0[8], p1[8];
#pragma unroll
    for (int i = 0; i < 8; ++i) {
      int kvA = kv0 + i + half * 8;
      float sA = st[0][i] * rsc;
      float sB = st[1][i] * rsc;
      if (kvA > qg)      sA = -1e30f;
      if (kvA + 16 > qg) sB = -1e30f;
      p0[i] = sA; p1[i] = sB;
      mloc = fmaxf(mloc, fmaxf(sA, sB));
    }
    mloc = fmaxf(mloc, __shfl_xor(mloc, 16, 32));
    float m_new = fmaxf(m_i, mloc);
    float alpha = __expf(m_i - m_new);
    float rs = 0.0f;
#pragma unroll
    for (int i = 0; i < 8; ++i) {
      p0[i] = __expf(p0[i] - m_new);
      p1[i] = __expf(p1[i] - m_new);
      rs += p0[i] + p1[i];
    }
    rs += __shfl_xor(rs, 16, 32);
    l_i = l_i * alpha + rs;
    m_i = m_new;

    // ---- pack P^T as B-operand: lane<16 needs kv 0..15 (own p0 + partner p0),
    //      lane>=16 needs kv 16..31 (partner p1 + own p1). One xor-16 shuffle.
    v16bf pb;
#pragma unroll
    for (int i = 0; i < 8; ++i) {
      float t0 = __shfl_xor(p0[i], 16, 32);
      float t1 = __shfl_xor(p1[i], 16, 32);
      float lo = half ? t1 : p0[i];
      float hi = half ? p1[i] : t0;
      pb[i]     = (__bf16)lo;
      pb[8 + i] = (__bf16)hi;
    }

    // ---- rescale O^T (per-lane: lane == q) and accumulate V^T * P^T
#pragma unroll
    for (int j = 0; j < 8; ++j)
#pragma unroll
      for (int e = 0; e < 8; ++e) o[j][e] *= alpha;
#pragma unroll
    for (int j = 0; j < 8; ++j) {
      v16bf av = *(const v16bf*)(Vh + (size_t)(j * 16 + ln) * SEQL + kv0 + half * 16);
      o[j] = wmma_bf16(av, pb, o[j]);
    }
  }

  // ---- finalize: O^T frag -> AttnO[q][h*128 + d] (bf16, GEMM-ready layout)
  const float invl = 1.0f / l_i;
#pragma unroll
  for (int j = 0; j < 8; ++j) {
#pragma unroll
    for (int i = 0; i < 8; ++i) {
      const int d = j * 16 + i + half * 8;
      AttnO[(size_t)qg * HID + (size_t)h * HDIM + d] = (__bf16)(o[j][i] * invl);
    }
  }
}

// =============================== launcher ==================================
extern "C" void kernel_launch(void* const* d_in, const int* in_sizes, int n_in,
                              void* d_out, int out_size, void* d_ws, size_t ws_size,
                              hipStream_t stream) {
  (void)in_sizes; (void)n_in; (void)out_size; (void)ws_size;
  const float* hidden   = (const float*)d_in[0];
  const float* residual = (const float*)d_in[1];
  const float* Wq       = (const float*)d_in[2];
  const float* Wk       = (const float*)d_in[3];
  const float* Wv       = (const float*)d_in[4];
  const float* Wo       = (const float*)d_in[5];
  // d_in[6] attention_mask / d_in[7] position_ids: causal + arange, computed in-kernel
  float* out = (float*)d_out;

  char* p = (char*)d_ws;
  __bf16* Hb  = (__bf16*)p; p += (size_t)SEQL * HID * 2;          // 16 MB
  __bf16* Wqb = (__bf16*)p; p += (size_t)HID  * HID * 2;          // 32 MB
  __bf16* Wkb = (__bf16*)p; p += (size_t)NKVH * HDIM * HID * 2;   //  8 MB
  __bf16* Wvb = (__bf16*)p; p += (size_t)NKVH * HDIM * HID * 2;   //  8 MB
  __bf16* Wob = (__bf16*)p; p += (size_t)HID  * HID * 2;          // 32 MB
  __bf16* Qb  = (__bf16*)p; p += (size_t)NHEAD * SEQL * HDIM * 2; // 16 MB
  __bf16* Kb  = (__bf16*)p; p += (size_t)NKVH  * SEQL * HDIM * 2; //  4 MB
  __bf16* Vt  = (__bf16*)p; p += (size_t)NKVH  * HDIM * SEQL * 2; //  4 MB
  __bf16* Ab  = (__bf16*)p;                                       // 16 MB

  auto cvt = [&](const float* src, __bf16* dst, size_t elems) {
    int n8 = (int)(elems / 8);
    cvt_f32_to_bf16<<<(n8 + 255) / 256, 256, 0, stream>>>(src, dst, n8);
  };
  auto gemmBlocks = [](int M, int N) {            // 128x128 tile per block
    return (M >> 7) * (N >> 7);
  };

  // 0) one-shot fp32 -> bf16 streaming conversion (keeps GEMM loops cvt-free)
  cvt(hidden, Hb,  (size_t)SEQL * HID);
  cvt(Wq,     Wqb, (size_t)HID * HID);
  cvt(Wk,     Wkb, (size_t)NKVH * HDIM * HID);
  cvt(Wv,     Wvb, (size_t)NKVH * HDIM * HID);
  cvt(Wo,     Wob, (size_t)HID * HID);

  // 1) QKV projections with fused epilogues (RoPE for Q/K, transpose for V)
  wmma_gemm_xwT<EPI_ROPE><<<gemmBlocks(SEQL, HID), 256, 0, stream>>>(
      Hb, Wqb, nullptr, (void*)Qb, SEQL, HID, HID);
  wmma_gemm_xwT<EPI_ROPE><<<gemmBlocks(SEQL, NKVH * HDIM), 256, 0, stream>>>(
      Hb, Wkb, nullptr, (void*)Kb, SEQL, NKVH * HDIM, HID);
  wmma_gemm_xwT<EPI_VT><<<gemmBlocks(SEQL, NKVH * HDIM), 256, 0, stream>>>(
      Hb, Wvb, nullptr, (void*)Vt, SEQL, NKVH * HDIM, HID);

  // 2) Causal flash attention (one wave per head x 16-row q block)
  {
    int waves = NHEAD * (SEQL / 16);
    flash_attn_wmma<<<(waves * 32 + 255) / 256, 256, 0, stream>>>(Qb, Kb, Vt, Ab);
  }

  // 3) o_proj + fused residual add -> fp32 output
  wmma_gemm_xwT<EPI_RES><<<gemmBlocks(SEQL, HID), 256, 0, stream>>>(
      Ab, Wob, residual, (void*)out, SEQL, HID, HID);
}
